// spatialAttention_45672682225720
// MI455X (gfx1250) — matrix operand order repaired and checked by reading, not compiled, not run on hardware
//
#include <hip/hip_runtime.h>

// ---------------------------------------------------------------------------
// spatialAttention fused pipeline for MI455X (gfx1250, wave32, WMMA bf16)
//
// Shapes: B=8, T=12, N=400, D=64, heads d=8, head-dim K=8.
// ~6 GFLOP of matmul vs ~530 MB of mandatory HBM traffic (491.5 MB of it the
// attn_out tensor) -> memory bound at 23.3 TB/s (~22 us floor).
// Strategy: bf16 WMMA (v_wmma_f32_16x16x32_bf16); weight matrices pre-
// converted once into WMMA-B-fragment-ordered bf16 (L2-resident, loaded as
// b128); all other global traffic coalesced b128; LDS staging for C-tile
// transposition; fused modulate/softmax in LDS; NT float4 attn_out stream.
// ---------------------------------------------------------------------------

typedef __attribute__((ext_vector_type(16))) __bf16       v16bf;
typedef __attribute__((ext_vector_type(8)))  __bf16       v8bf;
typedef __attribute__((ext_vector_type(8)))  float        v8f;
typedef __attribute__((ext_vector_type(4)))  float        f32x4;
typedef __attribute__((ext_vector_type(4)))  unsigned int u32x4;

#define Bdim 8
#define Tdim 12
#define Ndim 400
#define Ddim 64
#define HEADS 8
#define HK 8
#define ROWS (Bdim * Tdim * Ndim)          // 38400
#define Y_ELEMS ((size_t)ROWS * Ddim)      // 2457600

__device__ __forceinline__ v8f wmma_bf16(v16bf a, v16bf b, v8f c) {
  return __builtin_amdgcn_wmma_f32_16x16x32_bf16(
      /*neg_a=*/false, a, /*neg_b=*/false, b,
      /*c_mod=*/(short)0, c, /*reuse_a=*/false, /*reuse_b=*/false);
}

// A-fragment (16x32 bf16) from a row-major f32 tile, via 4x b128 loads.
// Layout (CDNA5 ISA): lanes0-15 row M=lane, K=k0+{0..7} then k0+{16..23};
// lanes16-31 same rows, K base +8.
__device__ __forceinline__ v16bf load_a_f32(const float* __restrict__ base,
                                            int ld, int k0, int lane) {
  const int r  = lane & 15;
  const int kb = k0 + ((lane & 16) ? 8 : 0);
  const float* p = base + r * ld + kb;
  const f32x4 q0 = *(const f32x4*)(p);
  const f32x4 q1 = *(const f32x4*)(p + 4);
  const f32x4 q2 = *(const f32x4*)(p + 16);
  const f32x4 q3 = *(const f32x4*)(p + 20);
  v16bf a;
#pragma unroll
  for (int e = 0; e < 4; ++e) {
    a[e]      = (__bf16)q0[e];
    a[e + 4]  = (__bf16)q1[e];
    a[e + 8]  = (__bf16)q2[e];
    a[e + 12] = (__bf16)q3[e];
  }
  return a;
}

// B-fragment (32x16 bf16) gathered from row-major f32 W (prep kernel only):
// lane%16 = column n, lanes0-15 K=k0+{0..7,16..23}; lanes16-31 K base +8.
__device__ __forceinline__ v16bf load_b_f32(const float* __restrict__ W,
                                            int ld, int k0, int n0, int lane) {
  const int n  = n0 + (lane & 15);
  const int kb = k0 + ((lane & 16) ? 8 : 0);
  v16bf b;
#pragma unroll
  for (int e = 0; e < 16; ++e) b[e] = (__bf16)W[(kb + (e < 8 ? e : e + 8)) * ld + n];
  return b;
}

// Fragment-ordered weight fetch: one 32B vector load (2x b128) per lane.
// frags layout: [matrix m][frag kc*4+ct][lane][16 bf16]
__device__ __forceinline__ v16bf load_wfrag(const __bf16* __restrict__ frags,
                                            int m, int kc, int ct, int lane) {
  return *(const v16bf*)(frags +
                         (((size_t)(m * 8 + kc * 4 + ct) * 32 + lane) * 16));
}

// ---------------------------------------------------------------------------
// Kernel 0: one-time conversion of the 5 weight matrices (Wq,Wk,Wv,W1,W2)
// into WMMA-B-fragment-ordered bf16 (40 KB total, stays in L2).
// ---------------------------------------------------------------------------
__global__ __launch_bounds__(32) void wprep_kernel(
    const float* __restrict__ Wq, const float* __restrict__ Wk,
    const float* __restrict__ Wv, const float* __restrict__ W1,
    const float* __restrict__ W2, __bf16* __restrict__ frags) {
  const int m    = blockIdx.x >> 3;
  const int fidx = blockIdx.x & 7;
  const int kc   = fidx >> 2;
  const int ct   = fidx & 3;
  const int lane = threadIdx.x;
  const float* W = (m == 0) ? Wq
                 : (m == 1) ? Wk
                 : (m == 2) ? Wv
                 : (m == 3) ? W1 : W2;
  const v16bf b = load_b_f32(W, Ddim, kc * 32, ct * 16, lane);
  *(v16bf*)(frags + (((size_t)(m * 8 + fidx) * 32 + lane) * 16)) = b;
}

// ---------------------------------------------------------------------------
// Kernel 1: S = softmax(relu(E @ E^T), axis=1) + lap       (400x400, f32)
// ---------------------------------------------------------------------------
__global__ __launch_bounds__(128) void supports_kernel(
    const float* __restrict__ emb, const float* __restrict__ lap,
    float* __restrict__ S) {
  const int i = blockIdx.x;
  __shared__ float vals[Ndim];
  __shared__ float red[128];
  float ei[10];
#pragma unroll
  for (int q = 0; q < 10; ++q) ei[q] = emb[i * 10 + q];

  float lmax = 0.f;  // relu >= 0
  for (int j = threadIdx.x; j < Ndim; j += blockDim.x) {
    float d = 0.f;
#pragma unroll
    for (int q = 0; q < 10; ++q) d += ei[q] * emb[j * 10 + q];
    d = fmaxf(d, 0.f);
    vals[j] = d;
    lmax = fmaxf(lmax, d);
  }
  red[threadIdx.x] = lmax;
  __syncthreads();
  for (int s = 64; s > 0; s >>= 1) {
    if ((int)threadIdx.x < s)
      red[threadIdx.x] = fmaxf(red[threadIdx.x], red[threadIdx.x + s]);
    __syncthreads();
  }
  const float m = red[0];
  __syncthreads();
  float lsum = 0.f;
  for (int j = threadIdx.x; j < Ndim; j += blockDim.x) {
    const float e = __expf(vals[j] - m);
    vals[j] = e;
    lsum += e;
  }
  red[threadIdx.x] = lsum;
  __syncthreads();
  for (int s = 64; s > 0; s >>= 1) {
    if ((int)threadIdx.x < s) red[threadIdx.x] += red[threadIdx.x + s];
    __syncthreads();
  }
  const float inv = 1.f / red[0];
  for (int j = threadIdx.x; j < Ndim; j += blockDim.x)
    S[i * Ndim + j] = vals[j] * inv + lap[i * Ndim + j];
}

// ---------------------------------------------------------------------------
// Kernel 2: q,k,v = x@W + b  (38400x64, K=64), emitted in bf16.
// One wave per 16-row tile; B-fragments fetched pre-packed (b128); C-tiles
// staged through LDS so global stores are 4x coalesced b128 per matrix.
// ---------------------------------------------------------------------------
__global__ __launch_bounds__(128) void qkv_kernel(
    const float* __restrict__ x, const __bf16* __restrict__ frags,
    const float* __restrict__ bq, const float* __restrict__ bk,
    const float* __restrict__ bv,
    __bf16* __restrict__ qb, __bf16* __restrict__ kb,
    __bf16* __restrict__ vb) {
  __shared__ __align__(16) __bf16 stage[4][16][Ddim];
  const int lane = threadIdx.x & 31;
  const int wave = threadIdx.x >> 5;
  const int row0 = (blockIdx.x * 4 + wave) * 16;
  const float* xrow = x + (size_t)row0 * Ddim;

  const v16bf a0 = load_a_f32(xrow, Ddim, 0, lane);
  const v16bf a1 = load_a_f32(xrow, Ddim, 32, lane);

  const float* bs[3] = {bq, bk, bv};
  __bf16* outs[3]    = {qb, kb, vb};

#pragma unroll
  for (int mtx = 0; mtx < 3; ++mtx) {
#pragma unroll
    for (int ct = 0; ct < 4; ++ct) {
      v8f c = {};
      c = wmma_bf16(a0, load_wfrag(frags, mtx, 0, ct, lane), c);
      c = wmma_bf16(a1, load_wfrag(frags, mtx, 1, ct, lane), c);
      const int col   = ct * 16 + (lane & 15);
      const float bia = bs[mtx][col];
      const int rl    = (lane & 16) ? 8 : 0;
#pragma unroll
      for (int r = 0; r < 8; ++r)
        stage[wave][rl + r][col] = (__bf16)(c[r] + bia);
    }
    // wave-local copy-out: 2 KB as 4x (ds_load_b128 + global_store_b128)
#pragma unroll
    for (int i = 0; i < 4; ++i) {
      const int f  = i * 256 + lane * 8;  // bf16 element index in tile
      const int rr = f >> 6;
      const int cc = f & 63;
      *(u32x4*)(outs[mtx] + (size_t)(row0 + rr) * Ddim + cc) =
          *(const u32x4*)&stage[wave][rr][cc];
    }
  }
}

// ---------------------------------------------------------------------------
// Kernel 3: fused attention.  grid = 768 pairs * 25 row-tiles.
//   attn = softmax((Q K^T / sqrt(8)) * S)  -> NT float4 stores (491.5 MB)
//   O    = attn @ V   (V-tile staged in LDS, 1/rowsum folded into A-frag)
// Head-dim 8 is zero-padded to K=32 for the bf16 WMMA; N padded 400->416.
// ---------------------------------------------------------------------------
__global__ __launch_bounds__(128) void attn_kernel(
    const __bf16* __restrict__ qb, const __bf16* __restrict__ kb,
    const __bf16* __restrict__ vb, const float* __restrict__ S,
    float* __restrict__ attn_out, float* __restrict__ O) {
  __shared__ __align__(16) float  attn[16][416];
  __shared__ float                red[16][8];
  __shared__ float                part[4][16][16];
  __shared__ __align__(16) __bf16 vtile[Ndim][HK];

  const int lane = threadIdx.x & 31;
  const int wave = threadIdx.x >> 5;

  const int pid     = blockIdx.x;
  const int rowTile = pid % 25;
  const int pr      = pid / 25;
  const int t       = pr % Tdim;
  const int hb      = pr / Tdim;  // hb = h*B + b
  const int h       = hb / Bdim;
  const int b       = hb % Bdim;

  // base into q/k/v/O [B,T,N,D] at (b,t,node=0,col=h*8)
  const size_t base = ((size_t)(b * Tdim + t) * Ndim) * Ddim + h * HK;
  const int n0 = rowTile * 16;

  // cooperative V-tile load: 400 rows x 16B, coalesced b128
  for (int node = threadIdx.x; node < Ndim; node += 128)
    *(u32x4*)&vtile[node][0] = *(const u32x4*)(vb + base + (size_t)node * Ddim);

  // zero the pad columns 400..415 of the LDS tile
  for (int idx = threadIdx.x; idx < 16 * 16; idx += 128)
    attn[idx >> 4][400 + (idx & 15)] = 0.f;

  // A-fragment: Q rows n0..n0+15, one b128 load per lane, padded 8->32
  v16bf aQ;
  {
    const int r = lane & 15;
    const v8bf qrow = *(const v8bf*)(qb + base + (size_t)(n0 + r) * Ddim);
    const bool lo = (lane & 16) == 0;
#pragma unroll
    for (int e = 0; e < 16; ++e)
      aQ[e] = (lo && e < HK) ? qrow[e & 7] : (__bf16)0.f;
  }

  for (int j = wave; j < 25; j += 4) {
    // B-fragment: K^T tile, column = key node, one b128 load per lane
    v16bf bK;
    {
      const int nn = j * 16 + (lane & 15);
      const v8bf krow = *(const v8bf*)(kb + base + (size_t)nn * Ddim);
      const bool lo = (lane & 16) == 0;
#pragma unroll
      for (int e = 0; e < 16; ++e)
        bK[e] = (lo && e < HK) ? krow[e & 7] : (__bf16)0.f;
    }
    v8f c = {};
    c = wmma_bf16(aQ, bK, c);
    const int col  = j * 16 + (lane & 15);
    const int rloc = (lane & 16) ? 8 : 0;
#pragma unroll
    for (int r = 0; r < 8; ++r)
      attn[rloc + r][col] = c[r];  // raw scores; modulate below
  }
  __syncthreads();

  // modulate: attn *= (S * 1/sqrt(d)), coalesced f32x4 loads of S
  {
    const float scale = 0.35355339059327373f;  // 1/sqrt(8)
    const float* Srow0 = S + (size_t)n0 * Ndim;
    for (int i = threadIdx.x; i < 16 * 100; i += 128) {
      const int rr = i / 100;
      const int cc = (i % 100) * 4;
      f32x4 s4 = *(const f32x4*)(Srow0 + rr * Ndim + cc);
      f32x4 a4 = *(const f32x4*)&attn[rr][cc];
      a4 = a4 * (s4 * scale);
      *(f32x4*)&attn[rr][cc] = a4;
    }
  }
  __syncthreads();

  // row softmax: 8 threads per row; LDS keeps raw exp values
  const int row = threadIdx.x >> 3;
  const int sub = threadIdx.x & 7;
  float m = -1e30f;
  for (int j = sub; j < Ndim; j += 8) m = fmaxf(m, attn[row][j]);
  red[row][sub] = m;
  __syncthreads();
  if (sub == 0) {
    float mm = red[row][0];
#pragma unroll
    for (int u = 1; u < 8; ++u) mm = fmaxf(mm, red[row][u]);
    red[row][0] = mm;
  }
  __syncthreads();
  m = red[row][0];
  __syncthreads();
  float s = 0.f;
  for (int j = sub; j < Ndim; j += 8) {
    const float e = __expf(attn[row][j] - m);
    attn[row][j] = e;
    s += e;
  }
  red[row][sub] = s;
  __syncthreads();
  if (sub == 0) {
    float ss = 0.f;
#pragma unroll
    for (int u = 0; u < 8; ++u) ss += red[row][u];
    red[row][0] = ss;
  }
  __syncthreads();

  // normalized attn -> attn_out [B, d*T, N, N], coalesced NT float4 bursts
  const float inv = 1.f / red[row][0];
  const size_t aobase =
      ((size_t)((b * HEADS + h) * Tdim + t)) * Ndim * Ndim +
      (size_t)(n0 + row) * Ndim;
  for (int j4 = sub; j4 < 100; j4 += 8) {
    f32x4 v4 = *(const f32x4*)&attn[row][j4 * 4];
    v4 = v4 * inv;
    __builtin_nontemporal_store(v4, (f32x4*)&attn_out[aobase + j4 * 4]);
  }

  // O-tile = attn(16x416, zero-pad) @ V(416x8 pad to x16); inv folded into A
  const float invA = 1.f / red[lane & 15][0];
  v8f acc = {};
  for (int kc = wave; kc < 13; kc += 4) {
    v16bf aA;
    {
      const int r = lane & 15;
#pragma unroll
      for (int e = 0; e < 16; ++e) {
        const int k = kc * 32 + ((lane & 16) ? 8 : 0) + (e < 8 ? e : e + 8);
        aA[e] = (__bf16)(attn[r][k] * invA);
      }
    }
    v16bf bV;
    {
      const int n   = lane & 15;
      const int kb0 = kc * 32 + ((lane & 16) ? 8 : 0);
#pragma unroll
      for (int e = 0; e < 16; ++e) {
        const int node = kb0 + (e < 8 ? e : e + 8);
        bV[e] = (n < HK && node < Ndim) ? vtile[node < Ndim ? node : 0][n]
                                        : (__bf16)0.f;
      }
    }
    acc = wmma_bf16(aA, bV, acc);
  }
  {
    const int rloc = (lane & 16) ? 8 : 0;
    const int col  = lane & 15;
#pragma unroll
    for (int r = 0; r < 8; ++r) part[wave][rloc + r][col] = acc[r];
  }
  __syncthreads();
  for (int idx = threadIdx.x; idx < 16 * HK; idx += 128) {
    const int r  = idx >> 3;
    const int kk = idx & 7;
    const float s4 =
        part[0][r][kk] + part[1][r][kk] + part[2][r][kk] + part[3][r][kk];
    O[base + (size_t)(n0 + r) * Ddim + kk] = s4;
  }
}

// ---------------------------------------------------------------------------
// Kernel 4: y = relu(O@W1+b1)@W2 + b2 + x   (two K=64 bf16 WMMA GEMMs)
// B-fragments pre-packed; y-tile staged in LDS; residual add fused into
// coalesced float4 copy-out.
// ---------------------------------------------------------------------------
__global__ __launch_bounds__(128) void ffn_kernel(
    const float* __restrict__ O, const float* __restrict__ x,
    const __bf16* __restrict__ frags, const float* __restrict__ b1,
    const float* __restrict__ b2, float* __restrict__ y) {
  __shared__ __align__(16) float hbuf[4][16][Ddim];
  const int lane = threadIdx.x & 31;
  const int wave = threadIdx.x >> 5;
  const int row0 = (blockIdx.x * 4 + wave) * 16;
  const float* orow = O + (size_t)row0 * Ddim;

  const v16bf a0 = load_a_f32(orow, Ddim, 0, lane);
  const v16bf a1 = load_a_f32(orow, Ddim, 32, lane);

#pragma unroll
  for (int ct = 0; ct < 4; ++ct) {
    v8f c = {};
    c = wmma_bf16(a0, load_wfrag(frags, 3, 0, ct, lane), c);
    c = wmma_bf16(a1, load_wfrag(frags, 3, 1, ct, lane), c);
    const int col = ct * 16 + (lane & 15);
    const float bia = b1[col];
    const int rl = (lane & 16) ? 8 : 0;
#pragma unroll
    for (int r = 0; r < 8; ++r)
      hbuf[wave][rl + r][col] = fmaxf(c[r] + bia, 0.f);
  }
  __syncthreads();

  const v16bf h0 = load_a_f32(&hbuf[wave][0][0], Ddim, 0, lane);
  const v16bf h1 = load_a_f32(&hbuf[wave][0][0], Ddim, 32, lane);

#pragma unroll
  for (int ct = 0; ct < 4; ++ct) {
    v8f c = {};
    c = wmma_bf16(h0, load_wfrag(frags, 4, 0, ct, lane), c);
    c = wmma_bf16(h1, load_wfrag(frags, 4, 1, ct, lane), c);
    const int col = ct * 16 + (lane & 15);
    const float bia = b2[col];
    const int rl = (lane & 16) ? 8 : 0;
#pragma unroll
    for (int r = 0; r < 8; ++r)
      hbuf[wave][rl + r][col] = c[r] + bia;  // overwrite, wave-local
  }
  // coalesced copy-out with fused residual: 4 KB as 8x float4 per lane
#pragma unroll
  for (int i = 0; i < 8; ++i) {
    const int f  = i * 128 + lane * 4;  // f32 element index in tile
    const int rr = f >> 6;
    const int cc = f & 63;
    const size_t gidx = (size_t)(row0 + rr) * Ddim + cc;
    f32x4 h4 = *(const f32x4*)&hbuf[wave][rr][cc];
    h4 = h4 + *(const f32x4*)(x + gidx);
    *(f32x4*)(y + gidx) = h4;
  }
}

// ---------------------------------------------------------------------------
// Launch. Workspace layout (bytes):
//   frags bf16 5*8*32*16      =    40,960
//   S     f32  400*400        =   640,000
//   qb    bf16 38400*64       = 4,915,200
//   kb    bf16 38400*64       = 4,915,200
//   vb    bf16 38400*64       = 4,915,200
//   O     f32  38400*64       = 9,830,400   (total ~25.3 MB)
// ---------------------------------------------------------------------------
extern "C" void kernel_launch(void* const* d_in, const int* in_sizes, int n_in,
                              void* d_out, int out_size, void* d_ws,
                              size_t ws_size, hipStream_t stream) {
  const float* x   = (const float*)d_in[0];
  const float* lap = (const float*)d_in[1];
  const float* emb = (const float*)d_in[2];
  const float* Wq  = (const float*)d_in[3];
  const float* bq  = (const float*)d_in[4];
  const float* Wk  = (const float*)d_in[5];
  const float* bk  = (const float*)d_in[6];
  const float* Wv  = (const float*)d_in[7];
  const float* bv  = (const float*)d_in[8];
  const float* W1  = (const float*)d_in[9];
  const float* b1  = (const float*)d_in[10];
  const float* W2  = (const float*)d_in[11];
  const float* b2  = (const float*)d_in[12];

  float* y        = (float*)d_out;
  float* attn_out = y + Y_ELEMS;

  char* ws      = (char*)d_ws;
  __bf16* frags = (__bf16*)ws;
  float*  S     = (float*)(ws + 40960);
  __bf16* qb    = (__bf16*)(ws + 40960 + 640000);
  __bf16* kb    = (__bf16*)(ws + 40960 + 640000 + 4915200);
  __bf16* vb    = (__bf16*)(ws + 40960 + 640000 + 2 * 4915200);
  float*  O     = (float*)(ws + 40960 + 640000 + 3 * 4915200);

  wprep_kernel<<<40, 32, 0, stream>>>(Wq, Wk, Wv, W1, W2, frags);
  supports_kernel<<<Ndim, 128, 0, stream>>>(emb, lap, S);
  qkv_kernel<<<ROWS / 64, 128, 0, stream>>>(x, frags, bq, bk, bv, qb, kb, vb);
  attn_kernel<<<HEADS * Bdim * Tdim * (Ndim / 16), 128, 0, stream>>>(
      qb, kb, vb, S, attn_out, O);
  ffn_kernel<<<ROWS / 64, 128, 0, stream>>>(O, x, frags, b1, b2, y);
}
